// MultiHeadAttention_17119739642246
// MI455X (gfx1250) — matrix operand order repaired and checked by reading, not compiled, not run on hardware
//
#include <hip/hip_runtime.h>
#include <hip/hip_bf16.h>

typedef _Float16 half_t;
typedef __attribute__((ext_vector_type(8)))  _Float16 v8h;
typedef __attribute__((ext_vector_type(16))) _Float16 v16h;
typedef __attribute__((ext_vector_type(8)))  float    v8f;
typedef int v4i __attribute__((vector_size(16)));

#define S_LEN 2048
#define BATCH 4
#define NHEAD 16
#define HDIM  64
#define DMODEL 1024
#define MTOT  (BATCH * S_LEN)   // 8192

#if defined(__has_builtin)
#if __has_builtin(__builtin_amdgcn_global_load_async_to_lds_b128) && \
    __has_builtin(__builtin_amdgcn_s_wait_asynccnt)
#define USE_ASYNC_LDS 1
#endif
#endif

__device__ __forceinline__ v16h make_v16h(v8h lo, v8h hi) {
  v16h r;
#pragma unroll
  for (int i = 0; i < 8; ++i) { r[i] = lo[i]; r[8 + i] = hi[i]; }
  return r;
}

__device__ __forceinline__ v8f vzero8f() {
  v8f z = {0.f, 0.f, 0.f, 0.f, 0.f, 0.f, 0.f, 0.f};
  return z;
}

// One 16-byte global -> LDS transfer (async DMA on gfx1250 when available).
__device__ __forceinline__ void g2l_b128(const half_t* g, half_t* l) {
#if defined(USE_ASYNC_LDS)
  __builtin_amdgcn_global_load_async_to_lds_b128(
      (__attribute__((address_space(1))) v4i*)(uintptr_t)g,
      (__attribute__((address_space(3))) v4i*)(uint32_t)(uintptr_t)l,
      0, 0);
#else
  *(v8h*)l = *(const v8h*)g;
#endif
}

__device__ __forceinline__ void wait_stage() {
#if defined(USE_ASYNC_LDS)
  __builtin_amdgcn_s_wait_asynccnt(0);
#endif
}

// ---------------------------------------------------------------------------
// fp32 -> f16 conversion (grid-stride)
// ---------------------------------------------------------------------------
__global__ void cvt_f32_to_f16(const float* __restrict__ src,
                               half_t* __restrict__ dst, int n) {
  int i = blockIdx.x * blockDim.x + threadIdx.x;
  int stride = gridDim.x * blockDim.x;
  for (; i < n; i += stride) dst[i] = (half_t)src[i];
}

// ---------------------------------------------------------------------------
// C[M,N] = A[M,K] * B[N,K]^T   (A,B f16; accumulate f32)
// mode 0: f16 out, row-major [M,N]
// mode 1: f16 out, V-transposed: out[b][col][sidx], b=row/2048, sidx=row%2048
// mode 2: f32 out + bias, row-major [M,N]
// Tile: BM=128, BN=64, BK=32. 256 threads = 8 waves; wave w -> rows w*16..+15,
// all 4 column tiles. LDS row stride 48 halfs (96B, 16B-aligned chunks).
// Double-buffered LDS; async global->LDS DMA overlapped with WMMA compute.
// ---------------------------------------------------------------------------
__global__ __launch_bounds__(256) void gemm_xwT(
    const half_t* __restrict__ A, const half_t* __restrict__ B,
    half_t* __restrict__ outh, float* __restrict__ outf,
    const float* __restrict__ bias, int mode) {
  __shared__ half_t As[2][128 * 48];
  __shared__ half_t Bs[2][64 * 48];
  const int K = DMODEL, N = DMODEL;
  const int m0 = blockIdx.y * 128;
  const int n0 = blockIdx.x * 64;
  const int tid = threadIdx.x;
  const int wave = tid >> 5, lane = tid & 31;
  const int r = lane & 15, g = lane >> 4;

  // Per-thread staging coordinates (each thread moves 3 x 16B per stage)
  const int ca = (tid & 3) * 8;          // k-column within tile (A chunks)
  const int ra0 = tid >> 2;              // A rows 0..63
  const int ra1 = (tid + 256) >> 2;      // A rows 64..127
  const int rb = tid >> 2, cb = (tid & 3) * 8;  // B row 0..63

  auto stage = [&](int buf, int kb) {
    g2l_b128(&A[(size_t)(m0 + ra0) * K + kb + ca], &As[buf][ra0 * 48 + ca]);
    g2l_b128(&A[(size_t)(m0 + ra1) * K + kb + ca], &As[buf][ra1 * 48 + ca]);
    g2l_b128(&B[(size_t)(n0 + rb) * K + kb + cb], &Bs[buf][rb * 48 + cb]);
  };

  v8f acc[4];
#pragma unroll
  for (int i = 0; i < 4; ++i) acc[i] = vzero8f();

  stage(0, 0);
  for (int kb = 0; kb < K; kb += 32) {
    const int buf = (kb >> 5) & 1;
    wait_stage();          // current buffer's DMA complete (this wave)
    __syncthreads();       // all waves' stages complete & visible
    if (kb + 32 < K) stage(buf ^ 1, kb + 32);  // overlap next DMA with WMMA

    // A fragment (16x32) for this wave's 16 rows
    v8h alo = *(const v8h*)&As[buf][(wave * 16 + r) * 48 + 8 * g];
    v8h ahi = *(const v8h*)&As[buf][(wave * 16 + r) * 48 + 8 * g + 16];
    v16h a = make_v16h(alo, ahi);
#pragma unroll
    for (int nt = 0; nt < 4; ++nt) {
      v8h blo = *(const v8h*)&Bs[buf][(nt * 16 + r) * 48 + 8 * g];
      v8h bhi = *(const v8h*)&Bs[buf][(nt * 16 + r) * 48 + 8 * g + 16];
      v16h bb = make_v16h(blo, bhi);
      acc[nt] = __builtin_amdgcn_wmma_f32_16x16x32_f16(
          false, a, false, bb, (short)0, acc[nt], false, false);
    }
  }

  // Epilogue: lane(r,g) vgpr j  ->  M = j + 8g, N = r
#pragma unroll
  for (int nt = 0; nt < 4; ++nt) {
#pragma unroll
    for (int j = 0; j < 8; ++j) {
      int row = m0 + wave * 16 + j + 8 * g;
      int col = n0 + nt * 16 + r;
      float v = acc[nt][j];
      if (mode == 0) {
        outh[(size_t)row * N + col] = (half_t)v;
      } else if (mode == 1) {
        int bidx = row >> 11, sidx = row & 2047;
        outh[(size_t)bidx * ((size_t)DMODEL * S_LEN) + (size_t)col * S_LEN +
             sidx] = (half_t)v;
      } else {
        outf[(size_t)row * N + col] = v + bias[col];
      }
    }
  }
}

// ---------------------------------------------------------------------------
// Flash attention: grid (s/64, b*h), block 128 (4 waves).
// Wave w handles 16 query rows. K block [64 keys x 64 d] and Vt block
// [64 d x 64 keys] double-buffered in LDS (stride 72 halfs = 144B).
// Online softmax: row stats replicated across each 16-lane group.
// ---------------------------------------------------------------------------
__global__ __launch_bounds__(128) void attn_flash(
    const half_t* __restrict__ Qh, const half_t* __restrict__ Kh,
    const half_t* __restrict__ Vt, half_t* __restrict__ Ctx) {
  __shared__ half_t Ks[2][64 * 72];
  __shared__ half_t Vs[2][64 * 72];
  __shared__ half_t Ps[64 * 72];

  const int bh = blockIdx.y;
  const int b = bh >> 4, h = bh & 15;
  const int q0 = blockIdx.x * 64;
  const half_t* Qb = Qh + (size_t)b * S_LEN * DMODEL + (size_t)h * HDIM;
  const half_t* Kb = Kh + (size_t)b * S_LEN * DMODEL + (size_t)h * HDIM;
  const half_t* Vb =
      Vt + (size_t)b * DMODEL * S_LEN + (size_t)h * HDIM * S_LEN;

  const int tid = threadIdx.x;
  const int wave = tid >> 5, lane = tid & 31;
  const int r = lane & 15, g = lane >> 4;

  auto stage_kv = [&](int buf, int kb) {
#pragma unroll
    for (int it = 0; it < 4; ++it) {
      int t = tid + it * 128;
      int row = t >> 3, col = (t & 7) * 8;
      g2l_b128(&Kb[(size_t)(kb * 64 + row) * DMODEL + col],
               &Ks[buf][row * 72 + col]);
      g2l_b128(&Vb[(size_t)row * S_LEN + kb * 64 + col],
               &Vs[buf][row * 72 + col]);
    }
  };

  // Q fragments (fixed for all kv blocks): A layout, row = r
  v16h qf[2];
  {
    int qrow = q0 + wave * 16 + r;
#pragma unroll
    for (int ks = 0; ks < 2; ++ks) {
      v8h lo = *(const v8h*)&Qb[(size_t)qrow * DMODEL + ks * 32 + 8 * g];
      v8h hi = *(const v8h*)&Qb[(size_t)qrow * DMODEL + ks * 32 + 8 * g + 16];
      qf[ks] = make_v16h(lo, hi);
    }
  }

  float m_i[8], l_i[8];
  v8f o[4];
#pragma unroll
  for (int j = 0; j < 8; ++j) { m_i[j] = -3.0e38f; l_i[j] = 0.f; }
#pragma unroll
  for (int nt = 0; nt < 4; ++nt) o[nt] = vzero8f();

  const int nkb = (int)blockIdx.x;  // last kv block index (causal)
  stage_kv(0, 0);
  for (int kb = 0; kb <= nkb; ++kb) {
    const int buf = kb & 1;
    wait_stage();
    __syncthreads();
    if (kb < nkb) stage_kv(buf ^ 1, kb + 1);  // overlap next K/V DMA

    // S = Q * K^T  (N = keys, K-dim = d = 64 -> 2 k-steps)
    v8f s[4];
#pragma unroll
    for (int nt = 0; nt < 4; ++nt) s[nt] = vzero8f();
#pragma unroll
    for (int nt = 0; nt < 4; ++nt) {
#pragma unroll
      for (int ks = 0; ks < 2; ++ks) {
        v8h blo =
            *(const v8h*)&Ks[buf][(nt * 16 + r) * 72 + ks * 32 + 8 * g];
        v8h bhi =
            *(const v8h*)&Ks[buf][(nt * 16 + r) * 72 + ks * 32 + 8 * g + 16];
        s[nt] = __builtin_amdgcn_wmma_f32_16x16x32_f16(
            false, qf[ks], false, make_v16h(blo, bhi), (short)0, s[nt], false,
            false);
      }
    }

    // Scale + causal mask + per-row block max
    float rmax[8];
#pragma unroll
    for (int j = 0; j < 8; ++j) rmax[j] = -3.0e38f;
#pragma unroll
    for (int nt = 0; nt < 4; ++nt) {
      int kglob = kb * 64 + nt * 16 + r;
#pragma unroll
      for (int j = 0; j < 8; ++j) {
        int qglob = q0 + wave * 16 + j + 8 * g;
        float v = s[nt][j] * 0.125f;  // 1/sqrt(64)
        if (kglob > qglob) v = -3.0e38f;
        s[nt][j] = v;
        rmax[j] = fmaxf(rmax[j], v);
      }
    }
    // Row lives across 16 lanes within each half-wave group
#pragma unroll
    for (int j = 0; j < 8; ++j)
#pragma unroll
      for (int msk = 1; msk < 16; msk <<= 1)
        rmax[j] = fmaxf(rmax[j], __shfl_xor(rmax[j], msk, 16));

    // Online softmax update; P -> LDS (f16, A-layout staging buffer)
#pragma unroll
    for (int j = 0; j < 8; ++j) {
      float mnew = fmaxf(m_i[j], rmax[j]);
      float alpha = __expf(m_i[j] - mnew);
      float sm = 0.f;
#pragma unroll
      for (int nt = 0; nt < 4; ++nt) {
        float p = __expf(s[nt][j] - mnew);
        sm += p;
        Ps[(wave * 16 + j + 8 * g) * 72 + nt * 16 + r] = (half_t)p;
      }
#pragma unroll
      for (int msk = 1; msk < 16; msk <<= 1)
        sm += __shfl_xor(sm, msk, 16);
      l_i[j] = l_i[j] * alpha + sm;
      m_i[j] = mnew;
#pragma unroll
      for (int nt = 0; nt < 4; ++nt) o[nt][j] *= alpha;
    }
    __syncthreads();  // make P visible for A-layout reload

    // O += P * V   (N = d, K-dim = keys = 64 -> 2 k-steps)
#pragma unroll
    for (int ks = 0; ks < 2; ++ks) {
      v8h plo = *(const v8h*)&Ps[(wave * 16 + r) * 72 + ks * 32 + 8 * g];
      v8h phi = *(const v8h*)&Ps[(wave * 16 + r) * 72 + ks * 32 + 8 * g + 16];
      v16h pa = make_v16h(plo, phi);
#pragma unroll
      for (int nt = 0; nt < 4; ++nt) {
        v8h blo =
            *(const v8h*)&Vs[buf][(nt * 16 + r) * 72 + ks * 32 + 8 * g];
        v8h bhi =
            *(const v8h*)&Vs[buf][(nt * 16 + r) * 72 + ks * 32 + 8 * g + 16];
        o[nt] = __builtin_amdgcn_wmma_f32_16x16x32_f16(
            false, pa, false, make_v16h(blo, bhi), (short)0, o[nt], false,
            false);
      }
    }
  }

  // Normalize and store ctx as f16 row-major [b*s, 1024]
#pragma unroll
  for (int j = 0; j < 8; ++j) {
    float inv = 1.0f / l_i[j];
    int row = q0 + wave * 16 + j + 8 * g;
#pragma unroll
    for (int nt = 0; nt < 4; ++nt) {
      Ctx[(size_t)(b * S_LEN + row) * DMODEL + h * HDIM + nt * 16 + r] =
          (half_t)(o[nt][j] * inv);
    }
  }
}

// ---------------------------------------------------------------------------
// Launch
// ---------------------------------------------------------------------------
extern "C" void kernel_launch(void* const* d_in, const int* in_sizes, int n_in,
                              void* d_out, int out_size, void* d_ws,
                              size_t ws_size, hipStream_t stream) {
  const float* x  = (const float*)d_in[0];
  const float* Wq = (const float*)d_in[1];
  const float* Wk = (const float*)d_in[2];
  const float* Wv = (const float*)d_in[3];
  const float* Wo = (const float*)d_in[4];
  const float* bo = (const float*)d_in[5];

  half_t* base = (half_t*)d_ws;
  size_t off = 0;
  half_t* Xh  = base + off; off += (size_t)MTOT * DMODEL;       // 16 MB
  half_t* Wqh = base + off; off += (size_t)DMODEL * DMODEL;     //  2 MB
  half_t* Wkh = base + off; off += (size_t)DMODEL * DMODEL;
  half_t* Wvh = base + off; off += (size_t)DMODEL * DMODEL;
  half_t* Woh = base + off; off += (size_t)DMODEL * DMODEL;
  half_t* Qh  = base + off; off += (size_t)MTOT * DMODEL;
  half_t* Kh  = base + off; off += (size_t)MTOT * DMODEL;
  half_t* Vth = base + off; off += (size_t)MTOT * DMODEL;       // [b,h,d,s]
  half_t* Ctx = base + off; off += (size_t)MTOT * DMODEL;

  // fp32 -> f16
  cvt_f32_to_f16<<<4096, 256, 0, stream>>>(x, Xh, MTOT * DMODEL);
  cvt_f32_to_f16<<<1024, 256, 0, stream>>>(Wq, Wqh, DMODEL * DMODEL);
  cvt_f32_to_f16<<<1024, 256, 0, stream>>>(Wk, Wkh, DMODEL * DMODEL);
  cvt_f32_to_f16<<<1024, 256, 0, stream>>>(Wv, Wvh, DMODEL * DMODEL);
  cvt_f32_to_f16<<<1024, 256, 0, stream>>>(Wo, Woh, DMODEL * DMODEL);

  // QKV projections (V stored transposed per head)
  dim3 ggrid(DMODEL / 64, MTOT / 128);
  gemm_xwT<<<ggrid, 256, 0, stream>>>(Xh, Wqh, Qh, nullptr, nullptr, 0);
  gemm_xwT<<<ggrid, 256, 0, stream>>>(Xh, Wkh, Kh, nullptr, nullptr, 0);
  gemm_xwT<<<ggrid, 256, 0, stream>>>(Xh, Wvh, Vth, nullptr, nullptr, 1);

  // Flash attention
  attn_flash<<<dim3(S_LEN / 64, BATCH * NHEAD), 128, 0, stream>>>(Qh, Kh, Vth,
                                                                  Ctx);

  // Output projection + bias -> fp32 d_out
  gemm_xwT<<<ggrid, 256, 0, stream>>>(Ctx, Woh, nullptr, (float*)d_out, bo, 2);
}
